// GCNTransform_72722386256375
// MI455X (gfx1250) — matrix operand order repaired and checked by reading, not compiled, not run on hardware
//
#include <hip/hip_runtime.h>
#include <hip/hip_bf16.h>
#include <math.h>

// ---------------------------------------------------------------------------
// CompGCN forward for MI455X (gfx1250, wave32).
// Edge phase: fp32 scatter-add of composed messages (memory-bound, ~2.3 GB).
// Dense phase: bf16 WMMA (v_wmma_f32_16x16x32_bf16), fused /3+bias+BN-stats.
// ---------------------------------------------------------------------------

typedef __attribute__((ext_vector_type(16))) __bf16 v16bf;
typedef __attribute__((ext_vector_type(8)))  __bf16 v8bf;
typedef __attribute__((ext_vector_type(8)))  float  v8f;

#define DD 128          // embedding dim (fixed by problem)
#define BN_EPS 1e-5f

// ------------------------------- degree ------------------------------------
__global__ void k_degree(const int* __restrict__ src, const int* __restrict__ dst,
                         float* deg_in, float* deg_out, int E) {
  int e = blockIdx.x * blockDim.x + threadIdx.x;
  if (e >= E) return;
  unsafeAtomicAdd(deg_in + src[e], 1.0f);
  unsafeAtomicAdd(deg_out + dst[e], 1.0f);
}

__global__ void k_rsqrt_inplace(float* a, float* b, int N) {
  int i = blockIdx.x * blockDim.x + threadIdx.x;
  if (i >= N) return;
  float x = a[i]; a[i] = (x > 0.f) ? rsqrtf(x) : 0.f;
  float y = b[i]; b[i] = (y > 0.f) ? rsqrtf(y) : 0.f;
}

// ---------------- weight transpose + f32->bf16 conversion -------------------
// wt[mat][n][k] = bf16( w_mat[k][n] )   (B operand wants column n contiguous in K)
__global__ void k_wconv(const float* __restrict__ w0, const float* __restrict__ w1,
                        const float* __restrict__ w2, __bf16* __restrict__ wt) {
  int idx = blockIdx.x * blockDim.x + threadIdx.x;
  if (idx >= 3 * DD * DD) return;
  int mat = idx >> 14;          // /16384
  int rem = idx & (DD * DD - 1);
  int n = rem >> 7;
  int k = rem & (DD - 1);
  const float* w = (mat == 0) ? w0 : (mat == 1) ? w1 : w2;
  wt[idx] = (__bf16)w[k * DD + n];
}

// ------------------------------- edge scatter -------------------------------
// One wave (32 lanes) per edge; each lane handles 4 contiguous dims (float4).
// agg_in[src]  += dinv_in[src]*dinv_in[dst]   * x[dst]*rel[etype]
// agg_out[dst] += dinv_out[src]*dinv_out[dst] * x[src]*rel[etype]
__global__ void k_scatter(const float* __restrict__ x, const float* __restrict__ rel,
                          const int* __restrict__ src, const int* __restrict__ dst,
                          const int* __restrict__ et,
                          const float* __restrict__ dinv_in, const float* __restrict__ dinv_out,
                          float* __restrict__ agg_in, float* __restrict__ agg_out, int E) {
  int wave = (blockIdx.x * blockDim.x + threadIdx.x) >> 5;
  int lane = threadIdx.x & 31;
  if (wave >= E) return;
  int s = src[wave], d = dst[wave], t = et[wave];
  float ci = dinv_in[s] * dinv_in[d];
  float co = dinv_out[s] * dinv_out[d];
  const float4 xs = *reinterpret_cast<const float4*>(x + (size_t)s * DD + lane * 4);
  const float4 xd = *reinterpret_cast<const float4*>(x + (size_t)d * DD + lane * 4);
  const float4 re = *reinterpret_cast<const float4*>(rel + (size_t)t * DD + lane * 4);
  float* ai = agg_in  + (size_t)s * DD + lane * 4;
  float* ao = agg_out + (size_t)d * DD + lane * 4;
  unsafeAtomicAdd(ai + 0, ci * xd.x * re.x);
  unsafeAtomicAdd(ai + 1, ci * xd.y * re.y);
  unsafeAtomicAdd(ai + 2, ci * xd.z * re.z);
  unsafeAtomicAdd(ai + 3, ci * xd.w * re.w);
  unsafeAtomicAdd(ao + 0, co * xs.x * re.x);
  unsafeAtomicAdd(ao + 1, co * xs.y * re.y);
  unsafeAtomicAdd(ao + 2, co * xs.z * re.z);
  unsafeAtomicAdd(ao + 3, co * xs.w * re.w);
}

// ------------------------------- WMMA helpers -------------------------------
// A fragment (16-bit A 16x32 layout): lane = half*16+row holds row `row`;
// elements 0..7  = K kb+half*8 .. +7 ; elements 8..15 = K kb+16+half*8 .. +7.
__device__ __forceinline__ v16bf frag_a(const float* p) {
  const float4 f0 = *reinterpret_cast<const float4*>(p);
  const float4 f1 = *reinterpret_cast<const float4*>(p + 4);
  const float4 f2 = *reinterpret_cast<const float4*>(p + 16);
  const float4 f3 = *reinterpret_cast<const float4*>(p + 20);
  v16bf a;
  a[0]=(__bf16)f0.x;  a[1]=(__bf16)f0.y;  a[2]=(__bf16)f0.z;  a[3]=(__bf16)f0.w;
  a[4]=(__bf16)f1.x;  a[5]=(__bf16)f1.y;  a[6]=(__bf16)f1.z;  a[7]=(__bf16)f1.w;
  a[8]=(__bf16)f2.x;  a[9]=(__bf16)f2.y;  a[10]=(__bf16)f2.z; a[11]=(__bf16)f2.w;
  a[12]=(__bf16)f3.x; a[13]=(__bf16)f3.y; a[14]=(__bf16)f3.z; a[15]=(__bf16)f3.w;
  return a;
}
// Same, with elementwise scale q[k] (self-loop compose x * loop_rel).
__device__ __forceinline__ v16bf frag_a_scaled(const float* p, const float* q) {
  const float4 f0 = *reinterpret_cast<const float4*>(p);
  const float4 f1 = *reinterpret_cast<const float4*>(p + 4);
  const float4 f2 = *reinterpret_cast<const float4*>(p + 16);
  const float4 f3 = *reinterpret_cast<const float4*>(p + 20);
  const float4 q0 = *reinterpret_cast<const float4*>(q);
  const float4 q1 = *reinterpret_cast<const float4*>(q + 4);
  const float4 q2 = *reinterpret_cast<const float4*>(q + 16);
  const float4 q3 = *reinterpret_cast<const float4*>(q + 20);
  v16bf a;
  a[0]=(__bf16)(f0.x*q0.x);  a[1]=(__bf16)(f0.y*q0.y);  a[2]=(__bf16)(f0.z*q0.z);  a[3]=(__bf16)(f0.w*q0.w);
  a[4]=(__bf16)(f1.x*q1.x);  a[5]=(__bf16)(f1.y*q1.y);  a[6]=(__bf16)(f1.z*q1.z);  a[7]=(__bf16)(f1.w*q1.w);
  a[8]=(__bf16)(f2.x*q2.x);  a[9]=(__bf16)(f2.y*q2.y);  a[10]=(__bf16)(f2.z*q2.z); a[11]=(__bf16)(f2.w*q2.w);
  a[12]=(__bf16)(f3.x*q3.x); a[13]=(__bf16)(f3.y*q3.y); a[14]=(__bf16)(f3.z*q3.z); a[15]=(__bf16)(f3.w*q3.w);
  return a;
}
// B fragment: lane = half*16+col holds column `col`; elements 0..7 = K kb+half*8 .. +7; 8..15 = K kb+16+half*8 .. +7.
__device__ __forceinline__ v16bf frag_b(const __bf16* p) {
  v8bf b0 = *reinterpret_cast<const v8bf*>(p);
  v8bf b1 = *reinterpret_cast<const v8bf*>(p + 16);
  v16bf b;
#pragma unroll
  for (int i = 0; i < 8; ++i) { b[i] = b0[i]; b[8 + i] = b1[i]; }
  return b;
}

// --------------------------------- GEMM -------------------------------------
// pre = (agg_in@Win + agg_out@Wout + (x*loop_rel)@Wloop)/3 + bias
// block = 8 waves; block b handles M-tile b, wave wv handles N-tile wv.
// Fused per-column sum / sum-of-squares for BatchNorm.
__global__ void k_gemm(const float* __restrict__ agg_in, const float* __restrict__ agg_out,
                       const float* __restrict__ xsrc,  const float* __restrict__ loop_rel_l,
                       const __bf16* __restrict__ wt,   const float* __restrict__ bias_l,
                       float* __restrict__ pre, float* col_sum, float* col_ssq, int tiles_m) {
  const int wv = threadIdx.x >> 5;
  const int lane = threadIdx.x & 31;
  const int wg = blockIdx.x * 8 + wv;
  const int mt = wg >> 3, nt = wg & 7;
  if (mt >= tiles_m) return;
  const int row = lane & 15, half = lane >> 4;
  const size_t arow = (size_t)(mt * 16 + row) * DD;

  v8f c = {};
#pragma unroll
  for (int mat = 0; mat < 3; ++mat) {
    const float* A = (mat == 0) ? agg_in : (mat == 1) ? agg_out : xsrc;
    const __bf16* Wr = wt + (size_t)mat * DD * DD + (size_t)(nt * 16 + row) * DD;
#pragma unroll
    for (int kb = 0; kb < DD; kb += 32) {
      const float* ap = A + arow + kb + half * 8;
      v16bf a = (mat == 2) ? frag_a_scaled(ap, loop_rel_l + kb + half * 8) : frag_a(ap);
      v16bf b = frag_b(Wr + kb + half * 8);
      c = __builtin_amdgcn_wmma_f32_16x16x32_bf16(false, a, false, b, (short)0, c, false, false);
    }
  }

  const int coln = nt * 16 + row;
  const float bval = bias_l[coln];
  float s = 0.f, ss = 0.f;
  const size_t obase = (size_t)(mt * 16 + half * 8) * DD + coln;
#pragma unroll
  for (int v = 0; v < 8; ++v) {
    float val = c[v] * (1.0f / 3.0f) + bval;
    pre[obase + (size_t)v * DD] = val;
    s += val; ss += val * val;
  }
  // lanes l and l^16 hold the same output column -> combine halves, then atomic
  s  += __shfl_xor(s, 16, 32);
  ss += __shfl_xor(ss, 16, 32);
  if (half == 0) {
    unsafeAtomicAdd(col_sum + coln, s);
    unsafeAtomicAdd(col_ssq + coln, ss);
  }
}

// ------------------------------ BatchNorm + tanh ----------------------------
__global__ void k_bnfin(const float* col_sum, const float* col_ssq,
                        float* mean, float* istd, float invN) {
  int j = threadIdx.x;
  if (j >= DD) return;
  float m = col_sum[j] * invN;
  float var = col_ssq[j] * invN - m * m;
  mean[j] = m;
  istd[j] = rsqrtf(var + BN_EPS);
}

__global__ void k_bntanh(const float* __restrict__ pre, const float* __restrict__ mean,
                         const float* __restrict__ istd, float* __restrict__ dst, size_t total) {
  size_t i = (size_t)blockIdx.x * blockDim.x + threadIdx.x;
  if (i >= total) return;
  int col = (int)(i & (DD - 1));
  dst[i] = tanhf((pre[i] - mean[col]) * istd[col]);
}

// ---------------------------- relation update -------------------------------
// out[i,:] = r[i,:] @ w   (i < R rows; (rel@w)[:-1] == rel[:-1]@w)
__global__ void k_relmm(const float* __restrict__ r, const float* __restrict__ w,
                        float* __restrict__ out) {
  const int i = blockIdx.x;
  const int j = threadIdx.x;
  __shared__ float rr[DD];
  rr[j] = r[(size_t)i * DD + j];
  __syncthreads();
  float acc = 0.f;
#pragma unroll 4
  for (int k = 0; k < DD; ++k) acc += rr[k] * w[(size_t)k * DD + j];
  out[(size_t)i * DD + j] = acc;
}

// --------------------------------- host -------------------------------------
extern "C" void kernel_launch(void* const* d_in, const int* in_sizes, int n_in,
                              void* d_out, int out_size, void* d_ws, size_t ws_size,
                              hipStream_t stream) {
  const float* node_emb = (const float*)d_in[0];
  const float* rel_emb  = (const float*)d_in[1];
  const float* w_in     = (const float*)d_in[2];
  const float* w_out    = (const float*)d_in[3];
  const float* w_loop   = (const float*)d_in[4];
  const float* w_rel    = (const float*)d_in[5];
  const float* loop_rel = (const float*)d_in[6];
  const float* bias     = (const float*)d_in[7];
  const int*   e_src    = (const int*)d_in[8];
  const int*   e_dst    = (const int*)d_in[9];
  const int*   e_typ    = (const int*)d_in[10];

  const int N = in_sizes[0] / DD;
  const int R = in_sizes[1] / DD;
  const int E = in_sizes[8];
  const int L = in_sizes[2] / (DD * DD);
  const size_t ND = (size_t)N * DD;
  const int tiles_m = N / 16;

  // workspace carve-out (256B aligned)
  char* ws = (char*)d_ws;
  size_t off = 0;
  auto carve = [&](size_t bytes) -> char* {
    char* p = ws + off;
    off += (bytes + 255) & ~(size_t)255;
    return p;
  };
  float*  dinv_in  = (float*)carve((size_t)N * 4);
  float*  dinv_out = (float*)carve((size_t)N * 4);
  float*  agg_in   = (float*)carve(ND * 4);
  float*  agg_out  = (float*)carve(ND * 4);
  float*  pre      = (float*)carve(ND * 4);
  float*  xbuf     = (float*)carve(ND * 4);
  float*  r1       = (float*)carve((size_t)R * DD * 4);
  float*  colsum   = (float*)carve(DD * 4);
  float*  colssq   = (float*)carve(DD * 4);
  float*  meanb    = (float*)carve(DD * 4);
  float*  istdb    = (float*)carve(DD * 4);
  __bf16* wt       = (__bf16*)carve((size_t)3 * DD * DD * 2);

  float* out_x = (float*)d_out;
  float* out_r = (float*)d_out + ND;

  // degrees (edge-structure only; shared by all layers)
  hipMemsetAsync(dinv_in, 0, (size_t)N * 4, stream);
  hipMemsetAsync(dinv_out, 0, (size_t)N * 4, stream);
  k_degree<<<(E + 255) / 256, 256, 0, stream>>>(e_src, e_dst, dinv_in, dinv_out, E);
  k_rsqrt_inplace<<<(N + 255) / 256, 256, 0, stream>>>(dinv_in, dinv_out, N);

  for (int l = 0; l < L; ++l) {
    const float* x_in = (l == 0) ? node_emb : xbuf;
    const float* r_in = (l == 0) ? rel_emb : r1;
    float* x_dst = (l == L - 1) ? out_x : xbuf;
    float* r_dst = (l == L - 1) ? out_r : r1;

    hipMemsetAsync(agg_in, 0, ND * 4, stream);
    hipMemsetAsync(agg_out, 0, ND * 4, stream);
    hipMemsetAsync(colsum, 0, DD * 4, stream);
    hipMemsetAsync(colssq, 0, DD * 4, stream);

    k_wconv<<<(3 * DD * DD + 255) / 256, 256, 0, stream>>>(
        w_in + (size_t)l * DD * DD, w_out + (size_t)l * DD * DD,
        w_loop + (size_t)l * DD * DD, wt);

    k_scatter<<<(E + 7) / 8, 256, 0, stream>>>(
        x_in, r_in, e_src, e_dst, e_typ, dinv_in, dinv_out, agg_in, agg_out, E);

    k_gemm<<<tiles_m, 256, 0, stream>>>(
        agg_in, agg_out, x_in, loop_rel + (size_t)l * DD, wt,
        bias + (size_t)l * DD, pre, colsum, colssq, tiles_m);

    k_bnfin<<<1, DD, 0, stream>>>(colsum, colssq, meanb, istdb, 1.0f / (float)N);
    k_bntanh<<<(unsigned)((ND + 255) / 256), 256, 0, stream>>>(pre, meanb, istdb, x_dst, ND);

    k_relmm<<<R, DD, 0, stream>>>(r_in, w_rel + (size_t)l * DD * DD, r_dst);
  }
}
